// GCNGraph_39728447488624
// MI455X (gfx1250) — compile-verified, matrix operand order checked by reading, not run on hardware
//
#include <hip/hip_runtime.h>

// ---------------------------------------------------------------------------
// Types for CDNA5 WMMA (wave32): V_WMMA_F32_16X16X32_F16
// ---------------------------------------------------------------------------
typedef __attribute__((ext_vector_type(16))) _Float16 v16h;
typedef __attribute__((ext_vector_type(8)))  float    v8f;

#define NEG_SLOPE 0.2f
#define HIDF 256

// ---------------------------------------------------------------------------
// Tiled WMMA GEMM: C[M,N] = A_op[M,K] @ B[K,N]   (f32 in/out, f16 WMMA, f32 acc)
//   TRANSA=0: A_op(m,k) = A[m*lda + k]
//   TRANSA=1: A_op(m,k) = A[k*lda + m]
//   IDENT=1 : A_op(m,k) += (m==k)           (implements A + I on the fly)
// Requires M%128==0, N%128==0, K%32==0 (true for all GEMMs in this model).
// Block = 256 threads = 8 waves; block tile 128x128x32.
// Wave grid 4x2; each wave owns a 32x64 sub-tile = 2x4 WMMA fragments.
// B tile is staged TRANSPOSED in LDS so both A and B fragment reads are
// 16 contiguous halves per lane -> ds_load_b128 (no scalar u16 gathers).
// All index math is 32-bit (max index 2048^2 < 2^31).
// ---------------------------------------------------------------------------
template <bool TRANSA, bool IDENT>
__global__ __launch_bounds__(256)
void gemm_wmma(const float* __restrict__ A, const float* __restrict__ B,
               float* __restrict__ C, int M, int N, int K,
               int lda, int ldb, int ldc)
{
    __shared__ _Float16 As[128][40];   // 128 x 32 (+8 pad)
    __shared__ _Float16 Bt[128][40];   // B transposed: [n][k], 128 x 32 (+8 pad)

    const int tid  = threadIdx.x;
    const int lane = tid & 31;
    const int w    = tid >> 5;
    const int wr   = w >> 1;          // wave row (0..3), 32 rows each
    const int wc   = w & 1;           // wave col (0..1), 64 cols each
    const int m0   = blockIdx.y * 128;
    const int n0   = blockIdx.x * 128;

    v8f acc[2][4];
#pragma unroll
    for (int i = 0; i < 2; ++i)
#pragma unroll
        for (int j = 0; j < 4; ++j)
#pragma unroll
            for (int e = 0; e < 8; ++e) acc[i][j][e] = 0.0f;

    for (int k0 = 0; k0 < K; k0 += 32) {
        // ---- stage A tile (128x32) into LDS as f16 ----
#pragma unroll
        for (int e = 0; e < 16; ++e) {
            int lin = e * 256 + tid;
            int r, c;
            if (!TRANSA) { r = lin >> 5;   c = lin & 31;  }  // col-fast: coalesced
            else         { r = lin & 127;  c = lin >> 7;  }  // row-fast: coalesced
            float v = TRANSA ? A[(k0 + c) * lda + (m0 + r)]
                             : A[(m0 + r) * lda + (k0 + c)];
            if (IDENT && (m0 + r) == (k0 + c)) v += 1.0f;
            As[r][c] = (_Float16)v;
        }
        // ---- stage B tile (32x128) transposed into LDS as f16 ----
#pragma unroll
        for (int e = 0; e < 16; ++e) {
            int lin = e * 256 + tid;
            int r = lin >> 7, c = lin & 127;                 // coalesced over c
            Bt[c][r] = (_Float16)B[(k0 + r) * ldb + (n0 + c)];
        }
        __syncthreads();

        const int kb = (lane >> 4) * 16;  // K-half owned by this lane group
        v16h af[2], bf[4];
#pragma unroll
        for (int mi = 0; mi < 2; ++mi) {
            int ar = wr * 32 + mi * 16 + (lane & 15);
#pragma unroll
            for (int e = 0; e < 16; ++e) af[mi][e] = As[ar][kb + e];
        }
#pragma unroll
        for (int ni = 0; ni < 4; ++ni) {
            int bc = wc * 64 + ni * 16 + (lane & 15);
#pragma unroll
            for (int e = 0; e < 16; ++e) bf[ni][e] = Bt[bc][kb + e];
        }
#pragma unroll
        for (int mi = 0; mi < 2; ++mi)
#pragma unroll
            for (int ni = 0; ni < 4; ++ni)
                acc[mi][ni] = __builtin_amdgcn_wmma_f32_16x16x32_f16(
                    false, af[mi], false, bf[ni],
                    (short)0, acc[mi][ni], false, false);
        __syncthreads();
    }

    // ---- write back (C/D layout: VGPR r -> row r + 8*(lane>=16), col = lane&15) ----
#pragma unroll
    for (int mi = 0; mi < 2; ++mi)
#pragma unroll
        for (int ni = 0; ni < 4; ++ni) {
            int col   = n0 + wc * 64 + ni * 16 + (lane & 15);
            int rbase = m0 + wr * 32 + mi * 16 + ((lane >= 16) ? 8 : 0);
#pragma unroll
            for (int r = 0; r < 8; ++r)
                C[(rbase + r) * ldc + col] = acc[mi][ni][r];
        }
}

// ---------------------------------------------------------------------------
// Elementwise / reduction helper kernels
// ---------------------------------------------------------------------------
__global__ void zero_f32(float* p, long long n) {
    long long i = (long long)blockIdx.x * blockDim.x + threadIdx.x;
    if (i < n) p[i] = 0.0f;
}

__global__ void scatter_edges(float* A, const int* ei, int E, int n) {
    int e = blockIdx.x * blockDim.x + threadIdx.x;
    if (e < E) A[ei[e] * n + ei[E + e]] = 1.0f;
}

// deg[j] = 1 + sum_i A[i][j];  dinv[j] = deg>0 ? rsqrt(deg) : 0
__global__ void col_deg_dinv(const float* A, float* dinv, int n) {
    int j = blockIdx.x * blockDim.x + threadIdx.x;
    if (j >= n) return;
    float s = 1.0f;
    for (int i = 0; i < n; ++i) s += A[i * n + j];
    dinv[j] = (s > 0.0f) ? rsqrtf(s) : 0.0f;
}

__global__ void scale_rows(const float* src, const float* sc, float* dst, int n, int f) {
    int idx = blockIdx.x * blockDim.x + threadIdx.x;
    if (idx < n * f) dst[idx] = src[idx] * sc[idx / f];
}

// in-place: g = relu(dinv[row]*g + b[col])
__global__ void gcn_epilogue(float* g, const float* dinv, const float* b, int n, int f) {
    int idx = blockIdx.x * blockDim.x + threadIdx.x;
    if (idx < n * f) {
        float v = dinv[idx / f] * g[idx] + b[idx % f];
        g[idx] = v > 0.0f ? v : 0.0f;
    }
}

__global__ void add_bias(float* x, const float* b, int n, int f) {
    int idx = blockIdx.x * blockDim.x + threadIdx.x;
    if (idx < n * f) x[idx] += b[idx % f];
}

// A[i][j] = (A[i][j]!=0 || i==j) ? 1 : 0   (raw adjacency -> mask_f, in place)
__global__ void binarize_inplace(float* A, int n) {
    int idx = blockIdx.x * blockDim.x + threadIdx.x;
    if (idx < n * n) {
        int i = idx / n, j = idx % n;
        A[idx] = (A[idx] != 0.0f || i == j) ? 1.0f : 0.0f;
    }
}

// out[j][t] = max over i with mask[i][j] of g[i][t]  (block per column j)
__global__ void masked_max_kernel(const float* maskf, const float* g, float* out, int n, int f) {
    int j = blockIdx.x, t = threadIdx.x;
    float m = -3.402823466e38f;
    for (int i = 0; i < n; ++i) {
        float mk = maskf[i * n + j];   // uniform within block -> broadcast
        if (mk != 0.0f) m = fmaxf(m, g[i * f + t]);
    }
    out[j * f + t] = m;
}

// out[row] = dot(X[row, 0:256], w) + (bias ? bias[0] : 0)   (block per row, 256 thr)
__global__ void vec_proj(const float* X, const float* w, const float* bias,
                         float* out, int f) {
    int row = blockIdx.x, t = threadIdx.x;
    __shared__ float red[256];
    red[t] = X[row * f + t] * w[t];
    __syncthreads();
    for (int st = 128; st > 0; st >>= 1) {
        if (t < st) red[t] += red[t + st];
        __syncthreads();
    }
    if (t == 0) out[row] = red[0] + (bias ? bias[0] : 0.0f);
}

// column-wise masked softmax of leaky_relu(v[i]+u[j]+attb); zeros where unmasked
__global__ void softmax_col(const float* maskf, const float* u, const float* v,
                            const float* attb, float* score, int n) {
    int j = blockIdx.x, t = threadIdx.x;
    const float uj = u[j] + attb[0];
    __shared__ float red[256];
    float mx = -3.402823466e38f;
    for (int i = t; i < n; i += 256) {
        if (maskf[i * n + j] != 0.0f) {
            float r = v[i] + uj;
            r = r > 0.0f ? r : NEG_SLOPE * r;
            mx = fmaxf(mx, r);
        }
    }
    red[t] = mx; __syncthreads();
    for (int st = 128; st > 0; st >>= 1) { if (t < st) red[t] = fmaxf(red[t], red[t + st]); __syncthreads(); }
    float cmax = red[0]; __syncthreads();

    float sum = 0.0f;
    for (int i = t; i < n; i += 256) {
        if (maskf[i * n + j] != 0.0f) {
            float r = v[i] + uj;
            r = r > 0.0f ? r : NEG_SLOPE * r;
            sum += expf(r - cmax);
        }
    }
    red[t] = sum; __syncthreads();
    for (int st = 128; st > 0; st >>= 1) { if (t < st) red[t] += red[t + st]; __syncthreads(); }
    float inv = 1.0f / red[0];

    for (int i = t; i < n; i += 256) {
        float mk = maskf[i * n + j];
        float r = v[i] + uj;
        r = r > 0.0f ? r : NEG_SLOPE * r;
        score[i * n + j] = (mk != 0.0f) ? expf(r - cmax) * inv : 0.0f;
    }
}

// fit[j] = sigmoid( sum_i mask[i][j]*a[i] - deg_in[j]*bb[j] + cc[j] )
__global__ void fitness_kernel(const float* maskf, const float* a, const float* bb,
                               const float* cc, float* fit, int n) {
    int j = blockIdx.x * blockDim.x + threadIdx.x;
    if (j >= n) return;
    float s = 0.0f, d = 0.0f;
    for (int i = 0; i < n; ++i) {
        float m = maskf[i * n + j];
        s += m * a[i];
        d += m;
    }
    float z = s - d * bb[j] + cc[j];
    fit[j] = 1.0f / (1.0f + expf(-z));
}

// deterministic top-k: rank by (value desc, index asc); matches jax.lax.top_k
__global__ void topk_rank(const float* fit, int n, int k, int* perm, float* fvals) {
    int j = blockIdx.x * blockDim.x + threadIdx.x;
    if (j >= n) return;
    float fj = fit[j];
    int r = 0;
    for (int i = 0; i < n; ++i) {
        float fi = fit[i];
        if (fi > fj || (fi == fj && i < j)) ++r;
    }
    if (r < k) { perm[r] = j; fvals[r] = fj; }
}

__global__ void gather_rows_scale(const float* xnew, const int* perm, const float* fvals,
                                  float* hout, int f) {
    int q = blockIdx.x, t = threadIdx.x;
    hout[q * f + t] = xnew[perm[q] * f + t] * fvals[q];
}

__global__ void gather_cols(const float* score, const int* perm, float* Sp, int n, int k) {
    int idx = blockIdx.x * blockDim.x + threadIdx.x;
    if (idx < n * k) {
        int i = idx / k, q = idx % k;
        Sp[idx] = score[i * n + perm[q]];
    }
}

__global__ void zero_diag(float* Ac, int k) {
    int q = blockIdx.x * blockDim.x + threadIdx.x;
    if (q < k) Ac[q * k + q] = 0.0f;
}

// ro[0:256] += mean over rows, ro[256:512] += max over rows   (one block, 256 thr)
__global__ void readout_accum(const float* h, float* ro, int k, int f) {
    int t = threadIdx.x;
    float s = 0.0f, m = -3.402823466e38f;
    for (int q = 0; q < k; ++q) {
        float v = h[q * f + t];
        s += v;
        m = fmaxf(m, v);
    }
    ro[t]     += s / (float)k;
    ro[f + t] += m;
}

// out = log_softmax( relu(ro @ W1 + b1) @ W2 + b2 )   (1x512 @ 512x256 ; 1x256 @ 256x10)
__global__ void final_mlp(const float* ro, const float* W1, const float* b1,
                          const float* W2, const float* b2, float* out) {
    __shared__ float o1[256];
    __shared__ float o2[16];
    int t = threadIdx.x;
    float s = b1[t];
    for (int r = 0; r < 512; ++r) s += ro[r] * W1[r * 256 + t];
    o1[t] = s > 0.0f ? s : 0.0f;
    __syncthreads();
    if (t < 10) {
        float s2 = b2[t];
        for (int c = 0; c < 256; ++c) s2 += o1[c] * W2[c * 10 + t];
        o2[t] = s2;
    }
    __syncthreads();
    if (t == 0) {
        float mx = o2[0];
        for (int i = 1; i < 10; ++i) mx = fmaxf(mx, o2[i]);
        float sum = 0.0f;
        for (int i = 0; i < 10; ++i) sum += expf(o2[i] - mx);
        float l = logf(sum) + mx;
        for (int i = 0; i < 10; ++i) out[i] = o2[i] - l;
    }
}

// ---------------------------------------------------------------------------
// Host orchestration
// ---------------------------------------------------------------------------
extern "C" void kernel_launch(void* const* d_in, const int* in_sizes, int n_in,
                              void* d_out, int out_size, void* d_ws, size_t ws_size,
                              hipStream_t stream) {
    const float* x      = (const float*)d_in[0];
    const int*   ei     = (const int*)  d_in[1];
    const float* W1     = (const float*)d_in[2];
    const float* b1     = (const float*)d_in[3];
    const float* W2     = (const float*)d_in[4];
    const float* b2     = (const float*)d_in[5];
    const float* linW   = (const float*)d_in[6];
    const float* linb   = (const float*)d_in[7];
    const float* attw   = (const float*)d_in[8];
    const float* attb   = (const float*)d_in[9];
    const float* le1W   = (const float*)d_in[10];
    const float* le1b   = (const float*)d_in[11];
    const float* le2W   = (const float*)d_in[12];
    const float* le3W   = (const float*)d_in[13];
    const float* le3b   = (const float*)d_in[14];
    const float* l1W    = (const float*)d_in[15];
    const float* l1b    = (const float*)d_in[16];
    const float* l2W    = (const float*)d_in[17];
    const float* l2b    = (const float*)d_in[18];

    const int N0 = 2048, F = HIDF;
    const int E = in_sizes[1] / 2;

    // bump allocator over workspace
    char* p = (char*)d_ws;
    auto alloc = [&](size_t bytes) -> void* {
        void* r = (void*)p;
        p += (bytes + 255) & ~(size_t)255;
        return r;
    };
    float* A     = (float*)alloc((size_t)N0 * N0 * 4);        // adjacency -> mask_f -> Ac
    float* score = (float*)alloc((size_t)N0 * N0 * 4);
    float* Sp    = (float*)alloc((size_t)N0 * (N0 / 2) * 4);
    float* T1    = (float*)alloc((size_t)N0 * (N0 / 2) * 4);
    float* tmp   = (float*)alloc((size_t)N0 * F * 4);
    float* tmp2  = (float*)alloc((size_t)N0 * F * 4);
    float* g     = (float*)alloc((size_t)N0 * F * 4);
    float* xmax  = (float*)alloc((size_t)N0 * F * 4);
    float* xq    = (float*)alloc((size_t)N0 * F * 4);
    float* xnew  = (float*)alloc((size_t)N0 * F * 4);
    float* hA    = (float*)alloc((size_t)N0 * F * 4);
    float* hB    = (float*)alloc((size_t)N0 * F * 4);
    float* dinv  = (float*)alloc((size_t)N0 * 4);
    float* u     = (float*)alloc((size_t)N0 * 4);
    float* v     = (float*)alloc((size_t)N0 * 4);
    float* avec  = (float*)alloc((size_t)N0 * 4);
    float* bvec  = (float*)alloc((size_t)N0 * 4);
    float* cvec  = (float*)alloc((size_t)N0 * 4);
    float* fit   = (float*)alloc((size_t)N0 * 4);
    float* fvals = (float*)alloc((size_t)N0 * 4);
    int*   perm  = (int*)  alloc((size_t)N0 * 4);
    float* ro    = (float*)alloc(512 * 4);

    // build dense adjacency; zero readout accumulator
    {
        long long nn = (long long)N0 * N0;
        zero_f32<<<(unsigned)((nn + 255) / 256), 256, 0, stream>>>(A, nn);
        scatter_edges<<<(E + 255) / 256, 256, 0, stream>>>(A, ei, E, N0);
        zero_f32<<<2, 256, 0, stream>>>(ro, 512);
    }

    const float* h = x;
    int n = N0, fin = 128;
    for (int lvl = 0; lvl < 3; ++lvl) {
        const float* W = (lvl == 0) ? W1 : W2;
        const float* b = (lvl == 0) ? b1 : b2;
        const int nf = n * F;

        // ---------- GCN: g = relu( dinv .* (A+I)^T @ (dinv .* (h@W)) + b ) ----------
        gemm_wmma<false, false><<<dim3(F / 128, n / 128), 256, 0, stream>>>(
            h, W, tmp, n, F, fin, fin, F, F);
        col_deg_dinv<<<(n + 255) / 256, 256, 0, stream>>>(A, dinv, n);
        scale_rows<<<(nf + 255) / 256, 256, 0, stream>>>(tmp, dinv, tmp2, n, F);
        gemm_wmma<true, true><<<dim3(F / 128, n / 128), 256, 0, stream>>>(
            A, tmp2, g, n, F, n, n, F, F);
        gcn_epilogue<<<(nf + 255) / 256, 256, 0, stream>>>(g, dinv, b, n, F);

        // ---------- ASAP pooling ----------
        binarize_inplace<<<(n * n + 255) / 256, 256, 0, stream>>>(A, n);  // A := mask_f

        masked_max_kernel<<<n, 256, 0, stream>>>(A, g, xmax, n, F);
        gemm_wmma<false, false><<<dim3(F / 128, n / 128), 256, 0, stream>>>(
            xmax, linW, xq, n, F, F, F, F, F);
        add_bias<<<(nf + 255) / 256, 256, 0, stream>>>(xq, linb, n, F);

        vec_proj<<<n, 256, 0, stream>>>(xq, attw,     nullptr, u, F);
        vec_proj<<<n, 256, 0, stream>>>(g,  attw + F, nullptr, v, F);
        softmax_col<<<n, 256, 0, stream>>>(A, u, v, attb, score, n);

        gemm_wmma<true, false><<<dim3(F / 128, n / 128), 256, 0, stream>>>(
            score, g, xnew, n, F, n, n, F, F);   // x_new = score^T @ g

        vec_proj<<<n, 256, 0, stream>>>(xnew, le1W, le1b,    avec, F);
        vec_proj<<<n, 256, 0, stream>>>(xnew, le2W, nullptr, bvec, F);
        vec_proj<<<n, 256, 0, stream>>>(xnew, le3W, le3b,    cvec, F);
        fitness_kernel<<<(n + 255) / 256, 256, 0, stream>>>(A, avec, bvec, cvec, fit, n);

        const int k2 = n / 2;
        topk_rank<<<(n + 255) / 256, 256, 0, stream>>>(fit, n, k2, perm, fvals);

        float* hnext = (lvl & 1) ? hB : hA;
        gather_rows_scale<<<k2, 256, 0, stream>>>(xnew, perm, fvals, hnext, F);

        gather_cols<<<(n * k2 + 255) / 256, 256, 0, stream>>>(score, perm, Sp, n, k2);

        // Ac = Sp^T @ (mask_f @ Sp); write Ac over A buffer (ld = k2)
        gemm_wmma<false, false><<<dim3(k2 / 128, n / 128), 256, 0, stream>>>(
            A, Sp, T1, n, k2, n, n, k2, k2);
        gemm_wmma<true, false><<<dim3(k2 / 128, k2 / 128), 256, 0, stream>>>(
            Sp, T1, A, k2, k2, n, k2, k2, k2);
        zero_diag<<<(k2 + 255) / 256, 256, 0, stream>>>(A, k2);

        readout_accum<<<1, 256, 0, stream>>>(hnext, ro, k2, F);

        h = hnext; n = k2; fin = F;
    }

    final_mlp<<<1, 256, 0, stream>>>(ro, l1W, l1b, l2W, l2b, (float*)d_out);
}
